// DSAttention_73555609911676
// MI455X (gfx1250) — matrix operand order repaired and checked by reading, not compiled, not run on hardware
//
#include <hip/hip_runtime.h>

// ---------------- CDNA5 WMMA fragment types ----------------
typedef __attribute__((ext_vector_type(16))) _Float16 v16h;
typedef __attribute__((ext_vector_type(8)))  float    v8f;
typedef __attribute__((ext_vector_type(4)))  _Float16 v4h;

union Frag16 {
    v16h v;
    _Float16 h[16];
    unsigned u[8];
};

// Problem constants (from reference setup_inputs)
#define BATCH 4
#define SEQ_L 2048
#define SEQ_S 2048
#define HEADS 8
#define DIM_E 64
#define DIM_D 64

// Tiling
#define QROWS_PER_BLOCK 128   // 8 waves x 16 rows
#define KTILE 64              // keys per inner tile (two WMMA K=32 steps for A*V)
#define PADROW 72             // padded LDS row (halves): 144B -> conflict-free 16-lane column reads

// ---------------- intra-16-lane row reductions (wave32) ----------------
__device__ __forceinline__ float rowmax16(float v) {
    v = fmaxf(v, __shfl_xor(v, 1, 32));
    v = fmaxf(v, __shfl_xor(v, 2, 32));
    v = fmaxf(v, __shfl_xor(v, 4, 32));
    v = fmaxf(v, __shfl_xor(v, 8, 32));
    return v;
}
__device__ __forceinline__ float rowsum16(float v) {
    v += __shfl_xor(v, 1, 32);
    v += __shfl_xor(v, 2, 32);
    v += __shfl_xor(v, 4, 32);
    v += __shfl_xor(v, 8, 32);
    return v;
}

// ======================================================================
// Pre-conversion kernels: fp32 -> f16 into workspace, attention-friendly
// layouts.  K16: [b][h][s][e] (tile rows contiguous); V16T: [b][h][d][s]
// (transposed so AV B-fragments read contiguous key pairs).
// ======================================================================
__global__ __launch_bounds__(256)
void convert_k_f16(const float* __restrict__ K, _Float16* __restrict__ K16) {
    const int t  = blockIdx.x * 256 + threadIdx.x;      // one float4 per thread
    const int e4 = (t & (DIM_E / 4 - 1)) << 2;
    int r = t >> 4;
    const int h = r & (HEADS - 1);  r >>= 3;
    const int s = r & (SEQ_S - 1);
    const int b = r >> 11;
    float4 f = *(const float4*)(K + ((size_t)(b * SEQ_S + s) * HEADS + h) * DIM_E + e4);
    v4h o = { (_Float16)f.x, (_Float16)f.y, (_Float16)f.z, (_Float16)f.w };
    *(v4h*)(K16 + ((size_t)(b * HEADS + h) * SEQ_S + s) * DIM_E + e4) = o;
}

__global__ __launch_bounds__(256)
void convert_v_f16t(const float* __restrict__ V, _Float16* __restrict__ V16T) {
    __shared__ _Float16 tile[32][DIM_D + 8];
    const int tid = threadIdx.x;
    const int b  = blockIdx.z, h = blockIdx.y;
    const int s0 = blockIdx.x * 32;
#pragma unroll
    for (int it = 0; it < 2; ++it) {
        const int idx = tid + it * 256;                 // 0..511
        const int sl  = idx >> 4;                       // 0..31
        const int d4  = (idx & 15) << 2;                // 0..60
        float4 f = *(const float4*)(V + ((size_t)(b * SEQ_S + s0 + sl) * HEADS + h) * DIM_D + d4);
        tile[sl][d4 + 0] = (_Float16)f.x;
        tile[sl][d4 + 1] = (_Float16)f.y;
        tile[sl][d4 + 2] = (_Float16)f.z;
        tile[sl][d4 + 3] = (_Float16)f.w;
    }
    __syncthreads();
#pragma unroll
    for (int it = 0; it < 2; ++it) {
        const int idx = tid + it * 256;                 // 0..511
        const int d   = idx >> 3;                       // 0..63
        const int s4  = (idx & 7) << 2;                 // 0..28
        v4h o = { tile[s4 + 0][d], tile[s4 + 1][d], tile[s4 + 2][d], tile[s4 + 3][d] };
        *(v4h*)(V16T + ((size_t)(b * HEADS + h) * DIM_D + d) * SEQ_S + s0 + s4) = o;
    }
}

// ======================================================================
// Per-tile compute, specialized on whether causal masking is needed.
// ======================================================================
template<bool MASKED>
__device__ __forceinline__ void process_tile(
        const _Float16 (*__restrict__ Ksh)[PADROW],    // [64 keys][64 e]
        const _Float16 (*__restrict__ VTsh)[PADROW],   // [64 d][64 keys]
        _Float16 (*__restrict__ Pw)[PADROW],           // [16 rows][64 keys]
        const Frag16& aQ0, const Frag16& aQ1,
        v8f* __restrict__ Oacc, float* __restrict__ mi, float* __restrict__ li,
        const float* __restrict__ deltaB, int jb, int q0,
        int nlo, int rowoff, int klaneoff, int alaneoff) {
    // ---------- S = (tau*scale*Q) * K^T : four 16-key column fragments ----------
    v8f s[4];
#pragma unroll
    for (int g = 0; g < 4; ++g) {
        Frag16 bK;
        const _Float16* krow = Ksh[g * 16 + nlo];
#pragma unroll
        for (int vv = 0; vv < 8; ++vv)
            bK.u[vv] = *(const unsigned*)&krow[klaneoff + 2 * vv];
        s[g] = (v8f){0.f,0.f,0.f,0.f,0.f,0.f,0.f,0.f};
        s[g] = __builtin_amdgcn_wmma_f32_16x16x32_f16(false, aQ0.v, false, bK.v,
                                                      (short)0, s[g], false, false);
#pragma unroll
        for (int vv = 0; vv < 8; ++vv)
            bK.u[vv] = *(const unsigned*)&krow[32 + klaneoff + 2 * vv];
        s[g] = __builtin_amdgcn_wmma_f32_16x16x32_f16(false, aQ1.v, false, bK.v,
                                                      (short)0, s[g], false, false);
    }

    // ---------- logits + online softmax over 64 columns ----------
    const int jA = jb + nlo, jB = jA + 16, jC = jA + 32, jD = jA + 48;
    const float dA = deltaB[jA] * 0.125f;
    const float dB = deltaB[jB] * 0.125f;
    const float dC = deltaB[jC] * 0.125f;
    const float dD = deltaB[jD] * 0.125f;

#pragma unroll
    for (int i = 0; i < 8; ++i) {
        float l0 = s[0][i] + dA;
        float l1 = s[1][i] + dB;
        float l2 = s[2][i] + dC;
        float l3 = s[3][i] + dD;
        if (MASKED) {
            const int qg = q0 + i + rowoff;
            if (jA > qg) l0 = -1.0e30f;
            if (jB > qg) l1 = -1.0e30f;
            if (jC > qg) l2 = -1.0e30f;
            if (jD > qg) l3 = -1.0e30f;
        }
        const float tmax  = rowmax16(fmaxf(fmaxf(l0, l1), fmaxf(l2, l3)));
        const float new_m = fmaxf(mi[i], tmax);
        const float corr  = __expf(mi[i] - new_m);
        mi[i] = new_m;

        const float p0 = __expf(l0 - new_m);
        const float p1 = __expf(l1 - new_m);
        const float p2 = __expf(l2 - new_m);
        const float p3 = __expf(l3 - new_m);
        li[i] = li[i] * corr + rowsum16((p0 + p1) + (p2 + p3));

#pragma unroll
        for (int nt = 0; nt < 4; ++nt) Oacc[nt][i] *= corr;

        _Float16* prow = Pw[i + rowoff];
        prow[nlo]      = (_Float16)p0;
        prow[16 + nlo] = (_Float16)p1;
        prow[32 + nlo] = (_Float16)p2;
        prow[48 + nlo] = (_Float16)p3;
    }

    // intra-wave cross-lane LDS RAW on P: drain the DS counter
    asm volatile("s_wait_dscnt 0x0" ::: "memory");

    // ---------- O += P * V : four 16-wide output column tiles ----------
    Frag16 aP0, aP1;
#pragma unroll
    for (int i = 0; i < 8; ++i) {
        const int k0 = alaneoff + (i << 1);
        aP0.u[i] = *(const unsigned*)&Pw[nlo][k0];
        aP1.u[i] = *(const unsigned*)&Pw[nlo][k0 + 32];
    }
#pragma unroll
    for (int nt = 0; nt < 4; ++nt) {
        Frag16 bV;
        const _Float16* vrow = VTsh[nt * 16 + nlo];
#pragma unroll
        for (int vv = 0; vv < 8; ++vv)
            bV.u[vv] = *(const unsigned*)&vrow[klaneoff + 2 * vv];
        Oacc[nt] = __builtin_amdgcn_wmma_f32_16x16x32_f16(false, aP0.v, false, bV.v,
                                                          (short)0, Oacc[nt], false, false);
#pragma unroll
        for (int vv = 0; vv < 8; ++vv)
            bV.u[vv] = *(const unsigned*)&vrow[32 + klaneoff + 2 * vv];
        Oacc[nt] = __builtin_amdgcn_wmma_f32_16x16x32_f16(false, aP1.v, false, bV.v,
                                                          (short)0, Oacc[nt], false, false);
    }
}

// ======================================================================
// Fast path: double-buffered async global->LDS copies of preconverted
// f16 K/V tiles (ASYNCcnt), WMMA flash attention.
// ======================================================================
__global__ __launch_bounds__(256)
void dsattn_flash_wmma_async(const float* __restrict__ Q,
                             const _Float16* __restrict__ K16,   // [b][h][s][e]
                             const _Float16* __restrict__ V16T,  // [b][h][d][s]
                             const float* __restrict__ tau,
                             const float* __restrict__ delta,
                             float* __restrict__ out) {
    __shared__ _Float16 KshD[2][KTILE][PADROW];     // 2 x 64 x 72 halves
    __shared__ _Float16 VTshD[2][DIM_D][PADROW];
    __shared__ _Float16 Psh[8][16][PADROW];

    const int tid  = threadIdx.x;
    const int lane = tid & 31;
    const int wid  = tid >> 5;

    const int b  = blockIdx.z;
    const int h  = blockIdx.y;
    const int qt = blockIdx.x;

    const int qbase = qt * QROWS_PER_BLOCK;
    const int q0    = qbase + wid * 16;
    const int q0s   = __builtin_amdgcn_readfirstlane(q0);  // wave-uniform scalar

    const int nlo      = lane & 15;
    const int grp      = lane >> 4;
    const int rowoff   = grp * 8;
    const int klaneoff = grp * 16;
    const int alaneoff = grp * 16;

    const int bh = b * HEADS + h;
    const unsigned long long kplane = (unsigned long long)(K16  + (size_t)bh * SEQ_S * DIM_E);
    const unsigned long long vplane = (unsigned long long)(V16T + (size_t)bh * DIM_D * SEQ_S);
    const float* deltaB = delta + (size_t)b * SEQ_S;

    const float a_tau = tau[b] * 0.125f;   // fold tau*rsqrt(E) into Q

    // ---------- this wave's Q stripe as two pre-scaled f16 A-fragments ----------
    Frag16 aQ0, aQ1;
    {
        const float* qp = Q + ((size_t)(b * SEQ_L + (q0 + nlo)) * HEADS + h) * DIM_E;
#pragma unroll
        for (int i = 0; i < 8; ++i) {
            const int k0 = alaneoff + (i << 1);
            float2 f0 = *(const float2*)(qp + k0);
            float2 f1 = *(const float2*)(qp + k0 + 32);
            aQ0.h[2 * i]     = (_Float16)(f0.x * a_tau);
            aQ0.h[2 * i + 1] = (_Float16)(f0.y * a_tau);
            aQ1.h[2 * i]     = (_Float16)(f1.x * a_tau);
            aQ1.h[2 * i + 1] = (_Float16)(f1.y * a_tau);
        }
    }

    // ---------- online-softmax state ----------
    float mi[8], li[8];
    v8f Oacc[4];
#pragma unroll
    for (int i = 0; i < 8; ++i) { mi[i] = -1.0e30f; li[i] = 0.0f; }
#pragma unroll
    for (int nt = 0; nt < 4; ++nt) Oacc[nt] = (v8f){0.f,0.f,0.f,0.f,0.f,0.f,0.f,0.f};

    const int ntiles = (qbase + QROWS_PER_BLOCK) / KTILE;   // uniform over block

    // per-iteration async tile issue: 2x B128 for K + 2x B128 for V per thread
    auto issue_tile = [&](int jb, int bufn) {
#pragma unroll
        for (int it = 0; it < 2; ++it) {
            const int t2  = tid + it * 256;                // 0..511
            const int row = t2 >> 3;                       // 0..63
            const int seg = t2 & 7;                        // 16B segment
            unsigned lds_k = (unsigned)(unsigned long long)&KshD[bufn][row][seg * 8];
            unsigned gof_k = (unsigned)(jb * DIM_E * 2 + t2 * 16);   // tile contiguous
            asm volatile("global_load_async_to_lds_b128 %0, %1, %2"
                         :: "v"(lds_k), "v"(gof_k), "s"(kplane) : "memory");
            unsigned lds_v = (unsigned)(unsigned long long)&VTshD[bufn][row][seg * 8];
            unsigned gof_v = (unsigned)(row * SEQ_S * 2 + jb * 2 + seg * 16);
            asm volatile("global_load_async_to_lds_b128 %0, %1, %2"
                         :: "v"(lds_v), "v"(gof_v), "s"(vplane) : "memory");
        }
    };

    issue_tile(0, 0);    // prologue

    for (int t = 0; t < ntiles; ++t) {
        const int jb = t * KTILE;
        if (t + 1 < ntiles) {
            issue_tile((t + 1) * KTILE, (t + 1) & 1);
            // 8 outstanding async ops; oldest 4 (this tile) must be done
            asm volatile("s_wait_asynccnt 0x4" ::: "memory");
        } else {
            asm volatile("s_wait_asynccnt 0x0" ::: "memory");
        }
        __syncthreads();   // all waves' copies for tile t complete

        if (jb <= q0s + 15) {          // wave-uniform: tile intersects causal region
            if (jb + KTILE - 1 <= q0s) // fully below diagonal: no masking
                process_tile<false>(KshD[t & 1], VTshD[t & 1], Psh[wid],
                                    aQ0, aQ1, Oacc, mi, li, deltaB, jb, q0,
                                    nlo, rowoff, klaneoff, alaneoff);
            else
                process_tile<true>(KshD[t & 1], VTshD[t & 1], Psh[wid],
                                   aQ0, aQ1, Oacc, mi, li, deltaB, jb, q0,
                                   nlo, rowoff, klaneoff, alaneoff);
        }
        __syncthreads();   // all waves done reading buf before it is overwritten
    }

    // ---------- normalize and write (B,L,H,D) fp32 ----------
#pragma unroll
    for (int i = 0; i < 8; ++i) {
        const float inv = 1.0f / li[i];
        const size_t orow = ((size_t)(b * SEQ_L + (q0 + i + rowoff)) * HEADS + h) * DIM_D;
#pragma unroll
        for (int nt = 0; nt < 4; ++nt)
            out[orow + nt * 16 + nlo] = Oacc[nt][i] * inv;
    }
}

// ======================================================================
// Fallback (round-1 kernel): fused fp32->f16 conversion in-kernel, used
// only if the workspace is too small for the preconverted copies.
// ======================================================================
__global__ __launch_bounds__(256)
void dsattn_flash_wmma_fb(const float* __restrict__ Q,
                          const float* __restrict__ K,
                          const float* __restrict__ V,
                          const float* __restrict__ tau,
                          const float* __restrict__ delta,
                          float* __restrict__ out) {
    __shared__ _Float16 Ksh[32][DIM_E + 8];
    __shared__ _Float16 VTsh[DIM_D][32 + 8];
    __shared__ _Float16 Psh[8][16][32 + 8];

    const int tid  = threadIdx.x;
    const int lane = tid & 31;
    const int wid  = tid >> 5;
    const int b  = blockIdx.z;
    const int h  = blockIdx.y;
    const int qbase = blockIdx.x * QROWS_PER_BLOCK;
    const int q0    = qbase + wid * 16;
    const int nlo      = lane & 15;
    const int grp      = lane >> 4;
    const int rowoff   = grp * 8;
    const int klaneoff = grp * 16;
    const int alaneoff = grp * 16;
    const float a_tau = tau[b] * 0.125f;

    Frag16 aQ0, aQ1;
    {
        const float* qp = Q + ((size_t)(b * SEQ_L + (q0 + nlo)) * HEADS + h) * DIM_E;
#pragma unroll
        for (int i = 0; i < 8; ++i) {
            const int k0 = alaneoff + (i << 1);
            float2 f0 = *(const float2*)(qp + k0);
            float2 f1 = *(const float2*)(qp + k0 + 32);
            aQ0.h[2 * i]     = (_Float16)f0.x;
            aQ0.h[2 * i + 1] = (_Float16)f0.y;
            aQ1.h[2 * i]     = (_Float16)f1.x;
            aQ1.h[2 * i + 1] = (_Float16)f1.y;
        }
    }

    float mi[8], li[8];
    v8f Oacc[4];
#pragma unroll
    for (int i = 0; i < 8; ++i) { mi[i] = -1.0e30f; li[i] = 0.0f; }
#pragma unroll
    for (int nt = 0; nt < 4; ++nt) Oacc[nt] = (v8f){0.f,0.f,0.f,0.f,0.f,0.f,0.f,0.f};

    const int jend = qbase + QROWS_PER_BLOCK;

    for (int jb = 0; jb < jend; jb += 32) {
        __syncthreads();
#pragma unroll
        for (int it = 0; it < 2; ++it) {
            const int idx = tid + it * 256;
            const int row = idx >> 4;
            const int c4  = (idx & 15) << 2;
            const size_t g = ((size_t)(b * SEQ_S + jb + row) * HEADS + h) * DIM_E + c4;
            float4 fk = *(const float4*)(K + g);
            v4h hk = { (_Float16)fk.x, (_Float16)fk.y, (_Float16)fk.z, (_Float16)fk.w };
            *(v4h*)&Ksh[row][c4] = hk;
            float4 fv = *(const float4*)(V + g);
            VTsh[c4 + 0][row] = (_Float16)fv.x;
            VTsh[c4 + 1][row] = (_Float16)fv.y;
            VTsh[c4 + 2][row] = (_Float16)fv.z;
            VTsh[c4 + 3][row] = (_Float16)fv.w;
        }
        __syncthreads();

        if (jb + 32 < jend) {
            const int prow = tid >> 3;
            const int pcol = (tid & 7) << 3;
            const size_t g = ((size_t)(b * SEQ_S + jb + 32 + prow) * HEADS + h) * DIM_E + pcol;
            __builtin_prefetch(K + g, 0, 3);
            __builtin_prefetch(V + g, 0, 3);
        }

        v8f s0 = (v8f){0.f,0.f,0.f,0.f,0.f,0.f,0.f,0.f};
        v8f s1 = (v8f){0.f,0.f,0.f,0.f,0.f,0.f,0.f,0.f};
        {
            Frag16 bK;
#pragma unroll
            for (int vv = 0; vv < 8; ++vv)
                bK.u[vv] = *(const unsigned*)&Ksh[nlo][klaneoff + 2 * vv];
            s0 = __builtin_amdgcn_wmma_f32_16x16x32_f16(false, aQ0.v, false, bK.v, (short)0, s0, false, false);
#pragma unroll
            for (int vv = 0; vv < 8; ++vv)
                bK.u[vv] = *(const unsigned*)&Ksh[nlo][32 + klaneoff + 2 * vv];
            s0 = __builtin_amdgcn_wmma_f32_16x16x32_f16(false, aQ1.v, false, bK.v, (short)0, s0, false, false);
#pragma unroll
            for (int vv = 0; vv < 8; ++vv)
                bK.u[vv] = *(const unsigned*)&Ksh[16 + nlo][klaneoff + 2 * vv];
            s1 = __builtin_amdgcn_wmma_f32_16x16x32_f16(false, aQ0.v, false, bK.v, (short)0, s1, false, false);
#pragma unroll
            for (int vv = 0; vv < 8; ++vv)
                bK.u[vv] = *(const unsigned*)&Ksh[16 + nlo][32 + klaneoff + 2 * vv];
            s1 = __builtin_amdgcn_wmma_f32_16x16x32_f16(false, aQ1.v, false, bK.v, (short)0, s1, false, false);
        }

        const int jA = jb + nlo;
        const int jB = jA + 16;
        const float dA = delta[(size_t)b * SEQ_S + jA] * 0.125f;
        const float dB = delta[(size_t)b * SEQ_S + jB] * 0.125f;

#pragma unroll
        for (int i = 0; i < 8; ++i) {
            const int qg = q0 + i + rowoff;
            float l0 = a_tau * s0[i] + dA;
            float l1 = a_tau * s1[i] + dB;
            if (jA > qg) l0 = -1.0e30f;
            if (jB > qg) l1 = -1.0e30f;
            const float tmax  = rowmax16(fmaxf(l0, l1));
            const float new_m = fmaxf(mi[i], tmax);
            const float corr  = __expf(mi[i] - new_m);
            mi[i] = new_m;
            const float p0 = __expf(l0 - new_m);
            const float p1 = __expf(l1 - new_m);
            li[i] = li[i] * corr + rowsum16(p0 + p1);
#pragma unroll
            for (int nt = 0; nt < 4; ++nt) Oacc[nt][i] *= corr;
            Psh[wid][i + rowoff][nlo]      = (_Float16)p0;
            Psh[wid][i + rowoff][16 + nlo] = (_Float16)p1;
        }

        asm volatile("s_wait_dscnt 0x0" ::: "memory");

        Frag16 aP;
#pragma unroll
        for (int i = 0; i < 8; ++i) {
            const int k0 = alaneoff + (i << 1);
            aP.u[i] = *(const unsigned*)&Psh[wid][nlo][k0];
        }
#pragma unroll
        for (int nt = 0; nt < 4; ++nt) {
            Frag16 bV;
#pragma unroll
            for (int vv = 0; vv < 8; ++vv)
                bV.u[vv] = *(const unsigned*)&VTsh[nt * 16 + nlo][klaneoff + 2 * vv];
            Oacc[nt] = __builtin_amdgcn_wmma_f32_16x16x32_f16(false, aP.v, false, bV.v, (short)0, Oacc[nt], false, false);
        }
    }

#pragma unroll
    for (int i = 0; i < 8; ++i) {
        const float inv = 1.0f / li[i];
        const size_t orow = ((size_t)(b * SEQ_L + (q0 + i + rowoff)) * HEADS + h) * DIM_D;
#pragma unroll
        for (int nt = 0; nt < 4; ++nt)
            out[orow + nt * 16 + nlo] = Oacc[nt][i] * inv;
    }
}

extern "C" void kernel_launch(void* const* d_in, const int* in_sizes, int n_in,
                              void* d_out, int out_size, void* d_ws, size_t ws_size,
                              hipStream_t stream) {
    const float* Q     = (const float*)d_in[0];  // (B,L,H,E)
    const float* K     = (const float*)d_in[1];  // (B,S,H,E)
    const float* V     = (const float*)d_in[2];  // (B,S,H,D)
    const float* tau   = (const float*)d_in[3];  // (B,1)
    const float* delta = (const float*)d_in[4];  // (B,S)
    float* out = (float*)d_out;                  // (B,L,H,D)

    const size_t halves_per_tensor = (size_t)BATCH * HEADS * SEQ_S * DIM_E; // 4.19M
    const size_t ws_need = halves_per_tensor * 2 * sizeof(_Float16);        // 16 MB

    dim3 grid(SEQ_L / QROWS_PER_BLOCK, HEADS, BATCH);  // (16, 8, 4)
    dim3 block(256);

    if (ws_size >= ws_need) {
        _Float16* K16  = (_Float16*)d_ws;
        _Float16* V16T = K16 + halves_per_tensor;
        convert_k_f16<<<dim3((BATCH * SEQ_S * HEADS * DIM_E / 4) / 256), block, 0, stream>>>(K, K16);
        convert_v_f16t<<<dim3(SEQ_S / 32, HEADS, BATCH), block, 0, stream>>>(V, V16T);
        dsattn_flash_wmma_async<<<grid, block, 0, stream>>>(Q, K16, V16T, tau, delta, out);
    } else {
        dsattn_flash_wmma_fb<<<grid, block, 0, stream>>>(Q, K, V, tau, delta, out);
    }
}
